// NeuralODE_16801912061971
// MI455X (gfx1250) — matrix operand order repaired and checked by reading, not compiled, not run on hardware
//
#include <hip/hip_runtime.h>

// ---------------------------------------------------------------- constants
constexpr int BB = 4096;   // batch
constexpr int DD = 512;    // state dim
constexpr int HH = 1024;   // hidden dim
constexpr int TT = 20;     // time points

// ---------------------------------------------------------------- types
typedef __attribute__((ext_vector_type(16))) _Float16 v16h;
typedef __attribute__((ext_vector_type(8)))  _Float16 v8h;
typedef __attribute__((ext_vector_type(4)))  _Float16 h4;
typedef __attribute__((ext_vector_type(8)))  float    v8f;

__device__ __forceinline__ float fast_tanh(float x) {
    x = fminf(fmaxf(x, -10.0f), 10.0f);
    float e = __expf(2.0f * x);
    return (e - 1.0f) / (e + 1.0f);
}

// build a 16x32 f16 A-fragment (contiguous k = 16*h + e order) from two contiguous 16B chunks
__device__ __forceinline__ v16h load_a16(const _Float16* p0, const _Float16* p1) {
    v8h lo = *(const v8h*)p0;
    v8h hi = *(const v8h*)p1;
    v16h a;
#pragma unroll
    for (int e = 0; e < 8; ++e) { a[e] = lo[e]; a[e + 8] = hi[e]; }
    return a;
}

// CDNA5 async global->LDS copy (16B per lane), tracked by ASYNCcnt.
// VDST = LDS byte address VGPR, VADDR = 64-bit global address (GV mode).
__device__ __forceinline__ void async_b128_to_lds(unsigned ldsoff, const _Float16* g) {
    asm volatile("global_load_async_to_lds_b128 %0, %1, off"
                 :: "v"(ldsoff), "v"(g)
                 : "memory");
}

// Pipeline barrier: wait only the counters the LDS double-buffer protocol
// needs (own async stores landed + own ds reads finished), then the split
// workgroup barrier. Crucially does NOT drain LOADcnt, so the A-prefetch
// global loads keep flowing across barriers.
__device__ __forceinline__ void pipe_barrier() {
    asm volatile(
        "s_wait_asynccnt 0\n\t"
        "s_wait_dscnt 0\n\t"
        "s_barrier_signal -1\n\t"
        "s_barrier_wait -1"
        ::: "memory");
}

// ---------------------------------------------------------------- weight prep
// W: [rows, cols] row-major f32.  Wt: [cols, rows] row-major f16 (transposed).
__global__ void node_transpose_f16(const float* __restrict__ W,
                                   _Float16* __restrict__ Wt,
                                   int rows, int cols) {
    int i = blockIdx.x * blockDim.x + threadIdx.x;   // linear over Wt
    int n = i / rows;
    int k = i % rows;
    Wt[i] = (_Float16)W[k * cols + n];
}

// y = x0 ; out[0] = x0
__global__ void node_init_y(const float* __restrict__ x0,
                            float* __restrict__ y,
                            float* __restrict__ out0) {
    int i = (blockIdx.x * blockDim.x + threadIdx.x) * 4;
    float4 v = *(const float4*)(x0 + i);
    *(float4*)(y + i)    = v;
    *(float4*)(out0 + i) = v;
}

// yin16 = f16(y + cmul*dt * kprev)
__global__ void node_make_yin(const float* __restrict__ y,
                              const float* __restrict__ kp,
                              _Float16* __restrict__ yin,
                              const float* __restrict__ tg,
                              int s, float cmul) {
    int i = (blockIdx.x * blockDim.x + threadIdx.x) * 4;
    float dt = tg[s + 1] - tg[s];
    float c = cmul * dt;
    float4 yv = *(const float4*)(y + i);
    h4 o;
    if (c != 0.0f) {
        float4 kv = *(const float4*)(kp + i);
        o[0] = (_Float16)fmaf(c, kv.x, yv.x);
        o[1] = (_Float16)fmaf(c, kv.y, yv.y);
        o[2] = (_Float16)fmaf(c, kv.z, yv.z);
        o[3] = (_Float16)fmaf(c, kv.w, yv.w);
    } else {
        o[0] = (_Float16)yv.x; o[1] = (_Float16)yv.y;
        o[2] = (_Float16)yv.z; o[3] = (_Float16)yv.w;
    }
    *(h4*)(yin + i) = o;
}

// y += dt/6 * acc ; out[t] = y
__global__ void node_update_y(float* __restrict__ y,
                              const float* __restrict__ acc,
                              float* __restrict__ outT,
                              const float* __restrict__ tg, int s) {
    int i = (blockIdx.x * blockDim.x + threadIdx.x) * 4;
    float dt = tg[s + 1] - tg[s];
    float sc = dt * (1.0f / 6.0f);
    float4 yv = *(const float4*)(y + i);
    float4 av = *(const float4*)(acc + i);
    yv.x = fmaf(sc, av.x, yv.x);  yv.y = fmaf(sc, av.y, yv.y);
    yv.z = fmaf(sc, av.z, yv.z);  yv.w = fmaf(sc, av.w, yv.w);
    *(float4*)(y + i)    = yv;
    *(float4*)(outT + i) = yv;
}

// ---------------------------------------------------------------- GEMM
// Block: 8 waves x (16M x 64N) strips stacked in M (block tile 128M x 64N).
// The 64x32 f16 weight chunk per K-step is staged into double-buffered LDS
// with GLOBAL_LOAD_ASYNC_TO_LDS_B128 (ASYNCcnt). Barriers drain only
// ASYNCcnt+DScnt, so the A-prefetch (LOADcnt) stream flows across them.
//
// A  [M,K] f16 row-major (activations), direct global loads.
// Bt [N,K] f16 row-major (pre-transposed weights), async-staged via LDS.
template <int Ndim, int Kdim, bool TANH>
__global__ __launch_bounds__(256) void node_gemm(
    const _Float16* __restrict__ A, const _Float16* __restrict__ Bt,
    const float* __restrict__ bias,
    _Float16* __restrict__ Hout,                       // TANH epilogue
    float* __restrict__ Kout, float* __restrict__ Acc, // RK4 epilogue
    float accscale, float wacc) {

    constexpr int LDSW = 40;               // halves per LDS row (80B, padded)
    constexpr int NC   = Kdim / 32;        // K chunks
    __shared__ _Float16 lsb[2][64 * LDSW];

    int tid = threadIdx.x;
    int lane = tid & 31, wave = tid >> 5;
    constexpr int NG = Ndim / 64;
    int m0 = (blockIdx.x / NG) * 128 + wave * 16;
    int n0 = (blockIdx.x % NG) * 64;
    int r = lane & 15, hs = lane >> 4;

    // A fragment source pointers (two 16B chunks per fragment)
    const _Float16* ap0 = A + (size_t)(m0 + r) * Kdim + hs * 16;
    const _Float16* ap1 = ap0 + 8;
    // B stage: thread t copies 16B: row t>>2 of the 64xK slab, 16B chunk t&3
    const _Float16* bsrc = Bt + (size_t)(n0 + (tid >> 2)) * Kdim + (tid & 3) * 8;
    const int stoff = (tid >> 2) * LDSW + (tid & 3) * 8;
    const int ldoff = r * LDSW + hs * 16;     // B fragment read base per lane

    // per-buffer LDS byte addresses for this thread's async store slot
    unsigned lds_st0 = (unsigned)(size_t)&lsb[0][stoff];
    unsigned lds_st1 = (unsigned)(size_t)&lsb[1][stoff];

    v8f acc[4] = {};

    // ---- pipeline prologue: async-stage chunk 0, load A chunk 0
    async_b128_to_lds(lds_st0, bsrc);
    v16h acur = load_a16(ap0, ap1);
    pipe_barrier();                           // lsb[0] valid for all waves

#pragma unroll
    for (int c = 0; c < NC; ++c) {
        // issue async stage of chunk c+1 (safe: prior barrier guarantees all
        // waves finished reading buffer (c+1)&1 in iteration c-1)
        if (c + 1 < NC)
            async_b128_to_lds(((c + 1) & 1) ? lds_st1 : lds_st0,
                              bsrc + (c + 1) * 32);
        // A prefetch for chunk c+1 (LOADcnt; settles at its WMMA, not at barrier)
        v16h anxt = acur;
        if (c + 1 < NC)
            anxt = load_a16(ap0 + (c + 1) * 32, ap1 + (c + 1) * 32);

        const _Float16* lb = &lsb[c & 1][ldoff];
        v16h b0 = *(const v16h*)(lb + 0 * 16 * LDSW);
        v16h b1 = *(const v16h*)(lb + 1 * 16 * LDSW);
        v16h b2 = *(const v16h*)(lb + 2 * 16 * LDSW);
        v16h b3 = *(const v16h*)(lb + 3 * 16 * LDSW);

        acc[0] = __builtin_amdgcn_wmma_f32_16x16x32_f16(false, acur, false, b0,
                                                        (short)0, acc[0], false, false);
        acc[1] = __builtin_amdgcn_wmma_f32_16x16x32_f16(false, acur, false, b1,
                                                        (short)0, acc[1], false, false);
        acc[2] = __builtin_amdgcn_wmma_f32_16x16x32_f16(false, acur, false, b2,
                                                        (short)0, acc[2], false, false);
        acc[3] = __builtin_amdgcn_wmma_f32_16x16x32_f16(false, acur, false, b3,
                                                        (short)0, acc[3], false, false);

        if (c + 1 < NC)
            pipe_barrier();                   // next buffer globally valid
        acur = anxt;
    }

    // ---- epilogue
#pragma unroll
    for (int j = 0; j < 4; ++j) {
        int ncol = n0 + j * 16 + r;
        float bv = bias[ncol];
        int rbase = m0 + hs * 8;
#pragma unroll
        for (int v = 0; v < 8; ++v) {
            float x = acc[j][v] + bv;
            if constexpr (TANH) {
                Hout[(size_t)(rbase + v) * Ndim + ncol] = (_Float16)fast_tanh(x);
            } else {
                size_t idx = (size_t)(rbase + v) * Ndim + ncol;
                Kout[idx] = x;
                Acc[idx] = accscale * Acc[idx] + wacc * x;
            }
        }
    }
}

// ---------------------------------------------------------------- launcher
extern "C" void kernel_launch(void* const* d_in, const int* in_sizes, int n_in,
                              void* d_out, int out_size, void* d_ws, size_t ws_size,
                              hipStream_t stream) {
    const float* x0 = (const float*)d_in[0];
    const float* tg = (const float*)d_in[1];
    const float* W1 = (const float*)d_in[2];
    const float* b1 = (const float*)d_in[3];
    const float* W2 = (const float*)d_in[4];
    const float* b2 = (const float*)d_in[5];
    float* out = (float*)d_out;

    // workspace carve-out
    float* y     = (float*)d_ws;                    // B*D f32
    float* kprev = y + BB * DD;                     // B*D f32
    float* acc   = kprev + BB * DD;                 // B*D f32
    _Float16* yin16 = (_Float16*)(acc + BB * DD);   // B*D f16
    _Float16* h16   = yin16 + BB * DD;              // B*H f16
    _Float16* W1t   = h16 + BB * HH;                // [H,D] f16 (W1^T)
    _Float16* W2t   = W1t + HH * DD;                // [D,H] f16 (W2^T)

    node_transpose_f16<<<(HH * DD) / 256, 256, 0, stream>>>(W1, W1t, DD, HH);
    node_transpose_f16<<<(DD * HH) / 256, 256, 0, stream>>>(W2, W2t, HH, DD);
    node_init_y<<<(BB * DD) / 1024, 256, 0, stream>>>(x0, y, out);

    const float cmul[4] = {0.0f, 0.5f, 0.5f, 1.0f};   // yin = y + cmul*dt*k_prev
    const float wacc[4] = {1.0f, 2.0f, 2.0f, 1.0f};   // RK4 weights

    const int grid_ew = (BB * DD) / 1024;
    const int grid_g1 = (BB / 128) * (HH / 64);       // 512 blocks
    const int grid_g2 = (BB / 128) * (DD / 64);       // 256 blocks

    for (int s = 0; s < TT - 1; ++s) {
        for (int st = 0; st < 4; ++st) {
            node_make_yin<<<grid_ew, 256, 0, stream>>>(y, kprev, yin16, tg, s, cmul[st]);
            node_gemm<HH, DD, true><<<grid_g1, 256, 0, stream>>>(
                yin16, W1t, b1, h16, nullptr, nullptr, 0.0f, 0.0f);
            node_gemm<DD, HH, false><<<grid_g2, 256, 0, stream>>>(
                h16, W2t, b2, nullptr, kprev, acc,
                (st == 0) ? 0.0f : 1.0f, wacc[st]);
        }
        node_update_y<<<grid_ew, 256, 0, stream>>>(
            y, acc, out + (size_t)(s + 1) * BB * DD, tg, s);
    }
}
